// GraphSAGE_4861902979554
// MI455X (gfx1250) — compile-verified, matrix operand order checked by reading, not run on hardware
//
#include <hip/hip_runtime.h>
#include <hip/hip_bf16.h>

// ---------- types for WMMA ----------
typedef __attribute__((ext_vector_type(16))) __bf16     v16bf;
typedef __attribute__((ext_vector_type(8)))  float        v8f;
typedef __attribute__((ext_vector_type(8)))  unsigned int v8u;

// vector type matching the async-LDS builtin's parameter: int __vector_size__(16)
typedef int vi4 __attribute__((__vector_size__(16)));

#define FEATS 256
#define KTOT  512   // fused K: [mean | x]

#define GLOBAL_AS __attribute__((address_space(1)))
#define LDS_AS    __attribute__((address_space(3)))

#if __has_builtin(__builtin_amdgcn_global_load_async_to_lds_b128)
#define HAVE_ASYNC_LDS 1
#else
#define HAVE_ASYNC_LDS 0
#endif

__device__ __forceinline__ void wait_async_zero() {
#if HAVE_ASYNC_LDS
#if __has_builtin(__builtin_amdgcn_s_wait_asynccnt)
    __builtin_amdgcn_s_wait_asynccnt(0);
#else
    asm volatile("s_wait_asynccnt 0x0" ::: "memory");
#endif
#endif
}

__device__ __forceinline__ unsigned short f2bf(float f) {
    unsigned int u = __float_as_uint(f);
    unsigned int r = u + 0x7FFFu + ((u >> 16) & 1u);   // round to nearest even
    return (unsigned short)(r >> 16);
}

// ---------- zero a float region ----------
__global__ void sage_zero(float* __restrict__ p, int n) {
    int i = blockIdx.x * blockDim.x + threadIdx.x;
    if (i < n) p[i] = 0.0f;
}

// ---------- pack [Wl ; Wr] (each HxK row-major fp32) into bf16 B[n][k], k in [0,512) ----------
__global__ void sage_pack_w(const float* __restrict__ Wl, const float* __restrict__ Wr,
                            unsigned short* __restrict__ Bcat) {
    int idx = blockIdx.x * blockDim.x + threadIdx.x;   // 256*512 total
    int n = idx >> 9;
    int k = idx & 511;
    float v = (k < FEATS) ? Wl[n * FEATS + k] : Wr[n * FEATS + (k - FEATS)];
    Bcat[idx] = f2bf(v);
}

// ---------- edge scatter: one wave32 per edge, mean-aggregation numerators + counts ----------
__global__ void sage_scatter(const float* __restrict__ xin,
                             const int* __restrict__ src, const int* __restrict__ dst,
                             float* __restrict__ agg, float* __restrict__ cnt, int nedges) {
    int e    = blockIdx.x * 8 + (threadIdx.x >> 5);
    int lane = threadIdx.x & 31;
    if (e >= nedges) return;
    int s = src[e];
    int d = dst[e];
    const float4* sp = (const float4*)(xin + (size_t)s * FEATS) + lane * 2;
    float4 a = sp[0];
    float4 b = sp[1];
    float* dp = agg + (size_t)d * FEATS + lane * 8;
    atomicAdd(dp + 0, a.x); atomicAdd(dp + 1, a.y);
    atomicAdd(dp + 2, a.z); atomicAdd(dp + 3, a.w);
    atomicAdd(dp + 4, b.x); atomicAdd(dp + 5, b.y);
    atomicAdd(dp + 6, b.z); atomicAdd(dp + 7, b.w);
    if (lane == 0) atomicAdd(cnt + d, 1.0f);
}

// ---------- fused SAGE GEMM: out = relu?([mean|x] @ Bcat^T + bias) ----------
// block: 256 thr = 8 waves; tile 64 rows x 128 cols; wave -> 16x64 (4 WMMA tiles)
__global__ void __launch_bounds__(256)
sage_gemm(const float* __restrict__ agg, const float* __restrict__ cnt,
          const float* __restrict__ xin, const unsigned short* __restrict__ Bcat,
          const float* __restrict__ bias, float* __restrict__ outp,
          int nrows, int relu) {
    const int LDA = 40;   // halfwords; 80 B rows -> 16 B aligned
    const int LDB = 40;
    __shared__ __attribute__((aligned(16))) unsigned short Alds[64 * LDA];
    __shared__ __attribute__((aligned(16))) unsigned short Blds[128 * LDB];

    int tid  = threadIdx.x;
    int wave = tid >> 5;
    int lane = tid & 31;
    int rg = wave & 3;        // row group (16 rows each)
    int cg = wave >> 2;       // col group 0/1 (64 cols each)
    int m  = lane & 15;
    int kh = lane >> 4;

    int rowBase = blockIdx.y * 64;
    int colBase = blockIdx.x * 128;

    // A staging: thread -> (row, 8 consecutive k)
    int arow = tid >> 2;                      // 0..63
    int akq  = (tid & 3) * 8;                 // 0,8,16,24
    int gr   = rowBase + arow;
    if (gr > nrows - 1) gr = nrows - 1;
    float c   = cnt[gr];
    float inv = 1.0f / fmaxf(c, 1.0f);

    // B staging: thread -> (col row of B, 16 consecutive k)
    int bn   = tid >> 1;                      // 0..127
    int bkq  = (tid & 1) * 16;
    int gcol = colBase + bn;                  // < 256 by grid

    v8f acc[4];
    v8f z = {0.f, 0.f, 0.f, 0.f, 0.f, 0.f, 0.f, 0.f};
#pragma unroll
    for (int t = 0; t < 4; ++t) acc[t] = z;

    for (int kstep = 0; kstep < KTOT; kstep += 32) {
        // ---- stage B tile (already bf16, [n][k] layout) ----
        // Pure copy: use gfx1250 async global->LDS path when available.
        {
            const unsigned short* gp = Bcat + (size_t)gcol * KTOT + kstep + bkq;
#if HAVE_ASYNC_LDS
            __builtin_amdgcn_global_load_async_to_lds_b128(
                (GLOBAL_AS vi4*)gp,
                (LDS_AS vi4*)&Blds[bn * LDB + bkq], 0, 0);
            __builtin_amdgcn_global_load_async_to_lds_b128(
                (GLOBAL_AS vi4*)(gp + 8),
                (LDS_AS vi4*)&Blds[bn * LDB + bkq + 8], 0, 0);
#else
            const uint4* wp = (const uint4*)gp;
            uint4 w0 = wp[0];
            uint4 w1 = wp[1];
            *(uint4*)&Blds[bn * LDB + bkq]     = w0;
            *(uint4*)&Blds[bn * LDB + bkq + 8] = w1;
#endif
        }
        // ---- stage A tile (fp32 -> bf16, mean division fused) ----
        {
            int kg = kstep + akq;
            const float* sp;
            float sc;
            if (kg < FEATS) { sp = agg + (size_t)gr * FEATS + kg;           sc = inv;  }
            else            { sp = xin + (size_t)gr * FEATS + (kg - FEATS); sc = 1.0f; }
            float4 f0 = ((const float4*)sp)[0];
            float4 f1 = ((const float4*)sp)[1];
            if (kstep + 32 < KTOT) __builtin_prefetch(sp + 32, 0, 1);  // global_prefetch_b8
            uint4 st;
            st.x = (unsigned int)f2bf(f0.x * sc) | ((unsigned int)f2bf(f0.y * sc) << 16);
            st.y = (unsigned int)f2bf(f0.z * sc) | ((unsigned int)f2bf(f0.w * sc) << 16);
            st.z = (unsigned int)f2bf(f1.x * sc) | ((unsigned int)f2bf(f1.y * sc) << 16);
            st.w = (unsigned int)f2bf(f1.z * sc) | ((unsigned int)f2bf(f1.w * sc) << 16);
            *(uint4*)&Alds[arow * LDA + akq] = st;
        }
        wait_async_zero();      // drain ASYNCcnt before the workgroup barrier
        __syncthreads();

        // ---- A fragment (16x32 bf16 per ISA layout) ----
        v8u au;
        const unsigned int* ap = (const unsigned int*)&Alds[(rg * 16 + m) * LDA];
#pragma unroll
        for (int j = 0; j < 8; ++j) {
            int kb = ((j < 4) ? (2 * j) : (16 + 2 * (j - 4))) + 8 * kh;
            au[j] = ap[kb >> 1];
        }
        v16bf af = __builtin_bit_cast(v16bf, au);

#pragma unroll
        for (int t = 0; t < 4; ++t) {
            const unsigned int* bp =
                (const unsigned int*)&Blds[(cg * 64 + t * 16 + m) * LDB];
            v8u bu;
#pragma unroll
            for (int j = 0; j < 8; ++j) {
                int kb = ((j < 4) ? (2 * j) : (16 + 2 * (j - 4))) + 8 * kh;
                bu[j] = bp[kb >> 1];
            }
            v16bf bfr = __builtin_bit_cast(v16bf, bu);
            acc[t] = __builtin_amdgcn_wmma_f32_16x16x32_bf16(
                false, af, false, bfr, (short)0, acc[t], false, false);
        }
        __syncthreads();
    }

    // ---- epilogue: bias (+relu), store fp32 ----
#pragma unroll
    for (int t = 0; t < 4; ++t) {
        int col = colBase + cg * 64 + t * 16 + m;     // N = lane&15
        float bb = bias[col];
#pragma unroll
        for (int r = 0; r < 8; ++r) {
            int row = rowBase + rg * 16 + r + 8 * kh; // M = r + 8*(lane>>4)
            if (row < nrows) {
                float v = acc[t][r] + bb;
                if (relu) v = fmaxf(v, 0.0f);
                outp[(size_t)row * FEATS + col] = v;
            }
        }
    }
}

extern "C" void kernel_launch(void* const* d_in, const int* in_sizes, int n_in,
                              void* d_out, int out_size, void* d_ws, size_t ws_size,
                              hipStream_t stream) {
    const float* x   = (const float*)d_in[0];
    const int*   ei  = (const int*)  d_in[1];
    const float* W1l = (const float*)d_in[2];
    const float* b1  = (const float*)d_in[3];
    const float* W1r = (const float*)d_in[4];
    const float* W2l = (const float*)d_in[5];
    const float* b2  = (const float*)d_in[6];
    const float* W2r = (const float*)d_in[7];
    float* out = (float*)d_out;

    int nnodes = in_sizes[0] / FEATS;
    int nedges = in_sizes[1] / 2;
    const int* src = ei;
    const int* dst = ei + nedges;

    // workspace carve-up
    float* agg = (float*)d_ws;                                   // nnodes*256
    float* cnt = agg + (size_t)nnodes * FEATS;                   // nnodes
    float* h   = cnt + nnodes;                                   // nnodes*256
    unsigned short* B1 = (unsigned short*)(h + (size_t)nnodes * FEATS);
    unsigned short* B2 = B1 + 256 * KTOT;

    int nz = nnodes * FEATS + nnodes;
    dim3 gemmGrid(2, (nnodes + 63) / 64);

    sage_pack_w<<<(256 * KTOT) / 256, 256, 0, stream>>>(W1l, W1r, B1);
    sage_pack_w<<<(256 * KTOT) / 256, 256, 0, stream>>>(W2l, W2r, B2);

    // ---- layer 1 ----
    sage_zero<<<(nz + 255) / 256, 256, 0, stream>>>(agg, nz);
    sage_scatter<<<(nedges + 7) / 8, 256, 0, stream>>>(x, src, dst, agg, cnt, nedges);
    sage_gemm<<<gemmGrid, 256, 0, stream>>>(agg, cnt, x, B1, b1, h, nnodes, 1);

    // ---- layer 2 ----
    sage_zero<<<(nz + 255) / 256, 256, 0, stream>>>(agg, nz);
    sage_scatter<<<(nedges + 7) / 8, 256, 0, stream>>>(h, src, dst, agg, cnt, nedges);
    sage_gemm<<<gemmGrid, 256, 0, stream>>>(agg, cnt, h, B2, b2, out, nnodes, 0);
}
